// FeatureHead_12979391169250
// MI455X (gfx1250) — matrix operand order repaired and checked by reading, not compiled, not run on hardware
//
#include <hip/hip_runtime.h>
#include <math.h>

typedef __attribute__((ext_vector_type(16))) _Float16 v16h;
typedef __attribute__((ext_vector_type(8)))  float    v8f;
typedef unsigned int u32x4 __attribute__((ext_vector_type(4)));
typedef int          i32x4 __attribute__((ext_vector_type(4)));
typedef int          i32x8 __attribute__((ext_vector_type(8)));

#define Hd 512
#define Wd 512
#define Bn 4
#define HW (Hd*Wd)
#define TS 16
#define HALO 3
#define IN_T 22          // TS + 2*HALO
#define OFF_OUT ((size_t)Bn*HW)                 // float offset of "out" (B,HW,5)
#define OFF_M   (OFF_OUT + (size_t)5*Bn*HW)    // float offset of mask m (B,HW)

#ifndef __has_builtin
#define __has_builtin(x) 0
#endif
#if __has_builtin(__builtin_amdgcn_global_load_async_to_lds_b32) && __has_builtin(__builtin_amdgcn_s_wait_asynccnt)
#define HAVE_ASYNC 1
#else
#define HAVE_ASYNC 0
#endif
#if __has_builtin(__builtin_amdgcn_tensor_load_to_lds) && __has_builtin(__builtin_amdgcn_s_wait_tensorcnt)
#define HAVE_TDM 1
#else
#define HAVE_TDM 0
#endif

// trapezoid weights [1,2,3,3,3,2,1]/15 = box5/5 (x) box3/3 per axis
__device__ __forceinline__ float trapval(int idx) {
    if (idx < 0 || idx > 6) return 0.f;
    int a = idx < 3 ? idx : 6 - idx;       // 0,1,2,3,2,1,0
    int v = a + 1; if (v > 3) v = 3;       // 1,2,3,3,3,2,1
    return (float)v * (1.f / 15.f);
}

#if HAVE_ASYNC
__device__ __forceinline__ void async_b32(const float* g, const float* l) {
    // per-lane LDS[l] = MEM[g]; builtin takes addrspace(1)/addrspace(3) int*.
    // AS casts via integer round-trip (generic-LDS low 32 bits are the LDS
    // byte offset per ISA 10.2 aperture rules)
    __builtin_amdgcn_global_load_async_to_lds_b32(
        (__attribute__((address_space(1))) int*)(unsigned long long)(const void*)g,
        (__attribute__((address_space(3))) int*)(unsigned int)(unsigned long long)(const void*)l,
        0, 0);
}
#endif

#if HAVE_TDM
// One 2D TDM descriptor: tile 22x22 of 4B elements, row stride 512 elements.
__device__ __forceinline__ void tdm_load_tile(unsigned long long gaddr, unsigned int lds_addr,
                                              unsigned int rem_x, unsigned int rem_y) {
    u32x4 g0;
    g0.x = 1u;                                                 // count=1, user descriptor
    g0.y = lds_addr;                                           // lds_addr [63:32]
    g0.z = (unsigned int)(gaddr & 0xFFFFFFFFull);              // global_addr[31:0]
    g0.w = (unsigned int)((gaddr >> 32) & 0x01FFFFFFull)       // global_addr[56:32]
         | (2u << 30);                                         // type=2 ("image")
    i32x8 g1 = {
        (int)(2u << 16),                                       // data_size=2 -> 4 bytes
        (int)((rem_x & 0xFFFFu) << 16),                        // tensor_dim0[15:0] @ bits 63:48
        (int)((rem_x >> 16) | ((rem_y & 0xFFFFu) << 16)),      // tensor_dim0[31:16] | tensor_dim1[15:0]
        (int)((rem_y >> 16) | ((unsigned)IN_T << 16)),         // tensor_dim1[31:16] | tile_dim0=22
        (int)IN_T,                                             // tile_dim1=22, tile_dim2=0
        (int)Wd,                                               // tensor_dim0_stride=512 (low)
        0, 0
    };
    i32x4 z4 = {0, 0, 0, 0};
#if __clang_major__ >= 23
    i32x8 z8 = {0, 0, 0, 0, 0, 0, 0, 0};
    __builtin_amdgcn_tensor_load_to_lds(g0, g1, z4, z4, z8, 0);
#else
    __builtin_amdgcn_tensor_load_to_lds(g0, g1, z4, z4, 0);
#endif
}
#endif

// ---------------- Stage 0: effective 7x7 per even channel: eff7 = box3/9 (*) w5
__global__ void k_prep(const float* __restrict__ wbbx, float* __restrict__ eff) {
    int i = blockIdx.x * blockDim.x + threadIdx.x;
    if (i >= 16 * 49) return;
    int ci = i / 49, d = i % 49, dy = d / 7, dx = d % 7;
    int c = 2 * ci;                        // even channels only (hmax is dead)
    int ky0 = dy >= 2 ? dy - 2 : 0, ky1 = dy < 4 ? dy : 4;
    int kx0 = dx >= 2 ? dx - 2 : 0, kx1 = dx < 4 ? dx : 4;
    float s = 0.f;
    for (int ky = ky0; ky <= ky1; ++ky)
        for (int kx = kx0; kx <= kx1; ++kx)
            s += wbbx[c * 25 + ky * 5 + kx];
    eff[i] = s * (1.f / 9.f);
}

// ---------------- Stage 1: fused tile kernel
__global__ __launch_bounds__(256) void k_main(const float* __restrict__ color,
                                              const float* __restrict__ wscore,
                                              const float* __restrict__ eff,
                                              float* __restrict__ dout) {
    __shared__ float   colS[16][IN_T][IN_T];  // even-channel halo tiles (31.0 KB)
    __shared__ float   tS[32][24];            // t tile (22x22 valid, rest zero)
    __shared__ _Float16 uS[32][16];           // intermediate U (x-pass result)
    __shared__ float   effS[16 * 49];
    __shared__ float   wsS[32];

    const int t  = threadIdx.x;
    const int b  = blockIdx.z;
    const int y0 = blockIdx.y * TS, x0 = blockIdx.x * TS;

    if (t < 32) wsS[t] = wscore[t];
    for (int i = t; i < 32 * 24; i += 256) ((float*)tS)[i] = 0.f;
    for (int i = t; i < 16 * 49; i += 256) effS[i] = eff[i];
    __syncthreads();

    // thread-owned halo positions (484 = 22*22 covered by e0 and e1)
    const int e0 = t, e1 = t + 256;
    const int r0 = e0 / IN_T, c0 = e0 % IN_T;
    const int r1 = e1 / IN_T, c1 = e1 % IN_T;
    const bool has1 = e1 < IN_T * IN_T;
    const int gy0 = y0 - HALO + r0, gx0 = x0 - HALO + c0;
    const int gy1 = y0 - HALO + r1, gx1 = x0 - HALO + c1;
    const bool ok0 = (gy0 >= 0) & (gy0 < Hd) & (gx0 >= 0) & (gx0 < Wd);
    const bool ok1 = has1 & (gy1 >= 0) & (gy1 < Hd) & (gx1 >= 0) & (gx1 < Wd);
    const float* cb = color + (size_t)b * 32 * HW;

    // ---- Phase 1: fill colS (even channels, TDM / async-to-LDS / plain)
    bool tdm_ok = false;
#if HAVE_TDM
    {
        const bool interior = (y0 >= HALO) && (x0 >= HALO) &&
                              (y0 + TS + HALO <= Hd) && (x0 + TS + HALO <= Wd);
        tdm_ok = interior;
        if (tdm_ok && t < 32) {            // one wave issues the 16 tensor DMAs
            const int gys = y0 - HALO, gxs = x0 - HALO;
            for (int ci = 0; ci < 16; ++ci) {
                const float* gp = cb + (size_t)(2 * ci) * HW + (size_t)gys * Wd + gxs;
                tdm_load_tile((unsigned long long)gp,
                              (unsigned int)(unsigned long long)(const void*)&colS[ci][0][0],
                              (unsigned int)(Wd - gxs), (unsigned int)(Hd - gys));
            }
            __builtin_amdgcn_s_wait_tensorcnt(0);
        }
    }
#endif
    if (!tdm_ok) {
#if HAVE_ASYNC
        // pre-zero OOB slots (disjoint from async destinations -> no race)
        if (!ok0)          for (int ci = 0; ci < 16; ++ci) colS[ci][r0][c0] = 0.f;
        if (has1 && !ok1)  for (int ci = 0; ci < 16; ++ci) colS[ci][r1][c1] = 0.f;
        if (ok0) for (int ci = 0; ci < 16; ++ci)
            async_b32(cb + (size_t)(2 * ci) * HW + gy0 * Wd + gx0, &colS[ci][r0][c0]);
        if (ok1) for (int ci = 0; ci < 16; ++ci)
            async_b32(cb + (size_t)(2 * ci) * HW + gy1 * Wd + gx1, &colS[ci][r1][c1]);
#else
        for (int c = 0; c < 32; c += 2) {
            const float* cp = cb + (size_t)c * HW;
            int ci = c >> 1;
            colS[ci][r0][c0] = ok0 ? cp[gy0 * Wd + gx0] : 0.f;
            if (has1) colS[ci][r1][c1] = ok1 ? cp[gy1 * Wd + gx1] : 0.f;
        }
#endif
    }
    // odd channels -> t partial sums (registers), overlapped with tile fill
    float tacc0 = 0.f, tacc1 = 0.f;
    for (int c = 1; c < 32; c += 2) {
        const float* cp = cb + (size_t)c * HW;
        if (c + 2 < 32 && ok0) __builtin_prefetch(cp + 2 * HW + gy0 * Wd + gx0, 0, 0);
        float w = wsS[c];
        if (ok0) tacc0 += w * cp[gy0 * Wd + gx0];
        if (ok1) tacc1 += w * cp[gy1 * Wd + gx1];
    }
#if HAVE_ASYNC
    if (!tdm_ok) __builtin_amdgcn_s_wait_asynccnt(0);
#endif
    __syncthreads();                       // colS visible to all waves

    // ---- Phase 2: even-channel t contributions from LDS, then t tile
    for (int ci = 0; ci < 16; ++ci) {
        float w = wsS[2 * ci];
        tacc0 += w * colS[ci][r0][c0];
        if (has1) tacc1 += w * colS[ci][r1][c1];
    }
    tS[r0][c0] = tacc0;
    if (has1) tS[r1][c1] = tacc1;
    __syncthreads();

    // ---- wmax: per-pixel 7x7 stencil over 16 even channels (VALU from LDS)
    const int px = t & 15, py = t >> 4;
    float wmax = -3.4e38f;
    for (int ci = 0; ci < 16; ++ci) {
        float acc = 0.f;
        const float* ef = &effS[ci * 49];
        #pragma unroll
        for (int dy = 0; dy < 7; ++dy) {
            #pragma unroll
            for (int dx = 0; dx < 7; ++dx)
                acc += ef[dy * 7 + dx] * colS[ci][py + dy][px + dx];
        }
        wmax = fmaxf(wmax, acc);
    }
    {   // stash wmax in out[..,1] (read-before-overwrite in k_out)
        size_t p = (size_t)(y0 + py) * Wd + (x0 + px);
        dout[OFF_OUT + ((size_t)b * HW + p) * 5 + 1] = wmax;
    }

    // ---- sc: separable trapezoid pooling as two matmuls on wave 0 (EXEC all-1s)
    if (t < 32) {
        const int l = t, m = l & 15, hi = l >> 4;
        v16h a1 = {}, a2 = {}, bx = {};
        #pragma unroll
        for (int e = 0; e < 16; ++e) {
            int v = e >> 1;
            int base = (v < 4) ? 2 * v : 16 + 2 * (v - 4);   // 16-bit A layout (ISA 7.12.2)
            int K = base + (hi ? 8 : 0) + (e & 1);
            a1[e] = (_Float16)((K < IN_T) ? tS[m][K] : 0.f);
            a2[e] = (_Float16)((K < IN_T) ? tS[16 + m][K] : 0.f);
            int Kb = ((e < 8) ? e : e + 8) + (hi ? 8 : 0);   // 16-bit B layout
            bx[e] = (_Float16)trapval(Kb - m);               // B[k][x]=trap[k-x]
        }
        v8f u1 = {}, u2 = {};
        u1 = __builtin_amdgcn_wmma_f32_16x16x32_f16(false, a1, false, bx, (short)0, u1, false, false);
        u2 = __builtin_amdgcn_wmma_f32_16x16x32_f16(false, a2, false, bx, (short)0, u2, false, false);
        #pragma unroll
        for (int v = 0; v < 8; ++v) {                        // C/D layout: M=v(+8), N=lane
            int M = v + (hi ? 8 : 0);
            uS[M][m]      = (_Float16)u1[v];
            uS[16 + M][m] = (_Float16)u2[v];
        }
    }
    __syncthreads();
    if (t < 32) {
        const int l = t, m = l & 15, hi = l >> 4;
        v16h ay = {}, bu = {};
        #pragma unroll
        for (int e = 0; e < 16; ++e) {
            int v = e >> 1;
            int base = (v < 4) ? 2 * v : 16 + 2 * (v - 4);
            int K = base + (hi ? 8 : 0) + (e & 1);
            ay[e] = (_Float16)trapval(K - m);                // A[y][k]=trap[k-y]
            int Kb = ((e < 8) ? e : e + 8) + (hi ? 8 : 0);
            bu[e] = uS[Kb][m];
        }
        v8f sc = {};
        sc = __builtin_amdgcn_wmma_f32_16x16x32_f16(false, ay, false, bu, (short)0, sc, false, false);
        #pragma unroll
        for (int v = 0; v < 8; ++v) {
            int row = y0 + v + (hi ? 8 : 0);
            int col = x0 + m;
            dout[(size_t)b * HW + (size_t)row * Wd + col] = sc[v];  // score_map
        }
    }
}

// ---------------- Stage 2: per-batch 10th-largest sc (sigmoid is monotone)
__global__ __launch_bounds__(256) void k_top(const float* __restrict__ dout,
                                             float* __restrict__ thr) {
    __shared__ float cand[256 * 10];
    __shared__ float red[256];
    __shared__ int   redi[256];
    __shared__ float kthS;
    const int b = blockIdx.x, t = threadIdx.x;
    const float* sm = dout + (size_t)b * HW;
    float top[10];
    #pragma unroll
    for (int i = 0; i < 10; ++i) top[i] = -3.4e38f;
    for (int i = t; i < HW; i += 256) {
        float v = sm[i];
        if (v > top[9]) {
            top[9] = v;
            #pragma unroll
            for (int j = 9; j > 0; --j)
                if (top[j] > top[j - 1]) { float tmp = top[j - 1]; top[j - 1] = top[j]; top[j] = tmp; }
        }
    }
    #pragma unroll
    for (int i = 0; i < 10; ++i) cand[t * 10 + i] = top[i];
    __syncthreads();
    for (int it = 0; it < 10; ++it) {
        float mv = -3.4e38f; int mi = -1;
        for (int i = t; i < 2560; i += 256) { float v = cand[i]; if (v > mv) { mv = v; mi = i; } }
        red[t] = mv; redi[t] = mi; __syncthreads();
        for (int s = 128; s > 0; s >>= 1) {
            if (t < s && (red[t + s] > red[t] || (red[t + s] == red[t] && redi[t + s] < redi[t]))) {
                red[t] = red[t + s]; redi[t] = redi[t + s];
            }
            __syncthreads();
        }
        if (t == 0) { kthS = red[0]; cand[redi[0]] = -3.4e38f; }
        __syncthreads();
    }
    if (t == 0) thr[b] = kthS;
}

// ---------------- Stage 3: final elementwise outputs (out tuple + mask m)
__global__ __launch_bounds__(256) void k_out(const float* __restrict__ thr,
                                             float* __restrict__ dout) {
    size_t gid = (size_t)blockIdx.x * 256 + threadIdx.x;
    if (gid >= (size_t)Bn * HW) return;
    int b = (int)(gid / HW);
    int p = (int)(gid % HW);
    float sc   = dout[(size_t)b * HW + p];
    float* op  = dout + OFF_OUT + gid * 5;
    float wmax = op[1];                               // stashed by k_main (read first)
    float s = 1.f / (1.f + expf(-sc));
    bool member = sc >= thr[b];                       // in top-10 (monotone in sc)
    bool m = (member && (s > 0.6f)) || (s > 0.9f);
    float xs = (float)(p % Wd), ys = (float)(p / Wd);
    float wv = expf(sc) * 10.f, hv = expf(wmax) * 10.f;
    float o0 = s, o1 = floorf(xs - wv), o2 = floorf(ys - hv),
          o3 = ceilf(xs + wv), o4 = ceilf(ys + hv);
    if (!m) { o0 = o1 = o2 = o3 = o4 = 0.f; }
    op[0] = o0; op[1] = o1; op[2] = o2; op[3] = o3; op[4] = o4;
    dout[OFF_M + gid] = m ? 1.f : 0.f;
}

extern "C" void kernel_launch(void* const* d_in, const int* in_sizes, int n_in,
                              void* d_out, int out_size, void* d_ws, size_t ws_size,
                              hipStream_t stream) {
    // d_in[0] = mask (unused by reference), d_in[1] = color, d_in[2] = w_bbx, d_in[3] = w_score
    const float* color  = (const float*)d_in[1];
    const float* wbbx   = (const float*)d_in[2];
    const float* wscore = (const float*)d_in[3];
    float* dout = (float*)d_out;
    float* eff  = (float*)d_ws;            // 784 floats
    float* thr  = (float*)d_ws + 1024;     // 4 floats

    k_prep<<<4, 256, 0, stream>>>(wbbx, eff);
    dim3 grid(Wd / TS, Hd / TS, Bn);       // 32 x 32 x 4 tiles
    k_main<<<grid, 256, 0, stream>>>(color, wscore, eff, dout);
    k_top<<<Bn, 256, 0, stream>>>(dout, thr);
    int n = Bn * HW;
    k_out<<<(n + 255) / 256, 256, 0, stream>>>(thr, dout);
}
